// StockPredictionModel_28432683499816
// MI455X (gfx1250) — matrix operand order repaired and checked by reading, not compiled, not run on hardware
//
#include <hip/hip_runtime.h>
#include <hip/hip_bf16.h>
#include <stdint.h>

// ---------------------------------------------------------------------------
// StockPredictionModel (xLSTM-ish) for MI455X / gfx1250, wave32.
// B=16, S=2048, D=256, L=4, 4D=1024.
// Pipeline per call:
//   prep_w    : Wx,Wh f32[L,D,4D] -> transposed bf16 [L,4D,D] (WMMA B layout)
//   in_proj   : feats @ in_W + in_b -> h f32 [B,S,D]
//   per layer : ln_cast (LN -> bf16 xn), gemm_gx (WMMA bf16 -> gx bf16, +bg),
//               scan_init (zero h-state + barrier), slstm_scan (sequential,
//               16 cooperating WGs, Wh in LDS via TDM, WMMA recurrent GEMM)
//   head      : h[:, -1, :] @ fc_W + fc_b -> d_out [B]
// Workspace use ~116 MB.
// ---------------------------------------------------------------------------

#define BB 16
#define SS 2048
#define DD 256
#define LL 4
#define FD 1024               // 4*D
#define MROWS (BB * SS)       // 32768

typedef __attribute__((ext_vector_type(16))) __bf16 v16bf;
typedef __attribute__((ext_vector_type(8)))  float  v8f;

typedef __attribute__((ext_vector_type(4))) unsigned int tdm_v4u;
typedef __attribute__((ext_vector_type(8))) int          tdm_v8i;
typedef __attribute__((ext_vector_type(4))) int          tdm_v4i;

union BPack { uint4 u[2]; v16bf v; };   // 32B = 8 VGPRs = one WMMA bf16 operand

// ---------------------------------------------------------------------------
// Weight prep: WxT[l][n][k] = bf16(Wx[l][k][n]); same for Wh. [L,4D,D] layout
// lets each WMMA B lane load contiguous K (32B) instead of a strided gather.
// ---------------------------------------------------------------------------
__global__ void prep_w_kernel(const float* __restrict__ Wx,
                              const float* __restrict__ Wh,
                              __hip_bfloat16* __restrict__ WxT,
                              __hip_bfloat16* __restrict__ WhT) {
  int idx = blockIdx.x * 256 + threadIdx.x;           // over L*4D*D = 1048576
  if (idx >= LL * FD * DD) return;
  int k = idx & (DD - 1);
  int n = (idx >> 8) & (FD - 1);
  int l = idx >> 18;
  size_t src = ((size_t)l * DD + k) * FD + n;
  WxT[idx] = __float2bfloat16(Wx[src]);
  WhT[idx] = __float2bfloat16(Wh[src]);
}

// ---------------------------------------------------------------------------
// Input projection: h[row,d] = in_b[d] + x[row]*W[0,d] + sum_k tf[row,k]*W[k+1,d]
// ---------------------------------------------------------------------------
__global__ void in_proj_kernel(const float* __restrict__ x,
                               const float* __restrict__ tf,
                               const float* __restrict__ W,
                               const float* __restrict__ bias,
                               float* __restrict__ h) {
  int idx = blockIdx.x * 256 + threadIdx.x;           // over B*S*D
  int d   = idx & (DD - 1);
  int row = idx >> 8;
  float acc = bias[d] + x[row] * W[d];
  const float* tfr = tf + (size_t)row * 6;
#pragma unroll
  for (int k = 0; k < 6; ++k) acc += tfr[k] * W[(k + 1) * DD + d];
  h[idx] = acc;
}

// ---------------------------------------------------------------------------
// LayerNorm row-wise over D=256, one wave per row (wave32: 8 elems/lane),
// output bf16 (WMMA A operand feed).
// ---------------------------------------------------------------------------
__global__ void ln_cast_kernel(const float* __restrict__ h,
                               const float* __restrict__ g,
                               const float* __restrict__ b,
                               __hip_bfloat16* __restrict__ xn) {
  int row  = blockIdx.x * 8 + (threadIdx.x >> 5);
  int lane = threadIdx.x & 31;
  const float* xr = h + (size_t)row * DD;
  float v[8], s = 0.f, s2 = 0.f;
#pragma unroll
  for (int j = 0; j < 8; ++j) {
    v[j] = xr[lane + 32 * j];
    s += v[j];
    s2 += v[j] * v[j];
  }
#pragma unroll
  for (int off = 16; off; off >>= 1) {
    s  += __shfl_xor(s, off, 32);
    s2 += __shfl_xor(s2, off, 32);
  }
  float mu  = s * (1.f / DD);
  float var = s2 * (1.f / DD) - mu * mu;
  float rs  = rsqrtf(var + 1e-5f);
#pragma unroll
  for (int j = 0; j < 8; ++j) {
    int d = lane + 32 * j;
    xn[(size_t)row * DD + d] = __float2bfloat16((v[j] - mu) * rs * g[d] + b[d]);
  }
}

// ---------------------------------------------------------------------------
// gx = xn @ Wx^T + bg   (M=32768, N=1024, K=256), bf16 in / f32 acc / bf16 out.
// WG = 8 waves; wave tile = 16(M) x 64(N); WG tile = 32(M) x 256(N).
// A layout (16-bit A 16x32): lanes 0-15 rows M, K 0..7 & 16..23;
//                            lanes 16-31 same rows, K 8..15 & 24..31.
// B from WxT[n][k]: lanes 0-15 col n, K 0..7 & 16..23; lanes 16-31 col n, K 8..15 & 24..31.
// ---------------------------------------------------------------------------
__global__ void __launch_bounds__(256, 2)
gemm_gx_kernel(const __hip_bfloat16* __restrict__ xn,   // [M, 256]
               const __hip_bfloat16* __restrict__ wT,   // [1024, 256]
               const float* __restrict__ bias,          // [1024]
               __hip_bfloat16* __restrict__ gx) {       // [M, 1024]
  const int lane = threadIdx.x & 31;
  const int wv   = threadIdx.x >> 5;
  const int wm   = wv >> 2;            // 0..1
  const int wn   = wv & 3;             // 0..3
  const int mBase = (blockIdx.x * 2 + wm) * 16;
  const int nBase = (blockIdx.y * 4 + wn) * 64;
  const int l15 = lane & 15, lh = lane >> 4;

  v8f acc[4];
#pragma unroll
  for (int t = 0; t < 4; ++t) acc[t] = (v8f){0, 0, 0, 0, 0, 0, 0, 0};

  const __hip_bfloat16* aRow = xn + (size_t)(mBase + l15) * DD + lh * 8;
#pragma unroll
  for (int kt = 0; kt < 8; ++kt) {
    const int k = kt * 32;
    BPack a;
    a.u[0] = *(const uint4*)(aRow + k);
    a.u[1] = *(const uint4*)(aRow + k + 16);
#pragma unroll
    for (int t = 0; t < 4; ++t) {
      const __hip_bfloat16* bRow =
          wT + (size_t)(nBase + t * 16 + l15) * DD + k + lh * 8;
      BPack bm;
      bm.u[0] = *(const uint4*)(bRow);
      bm.u[1] = *(const uint4*)(bRow + 16);
      acc[t] = __builtin_amdgcn_wmma_f32_16x16x32_bf16(
          false, a.v, false, bm.v, (short)0, acc[t], false, false);
    }
  }
#pragma unroll
  for (int t = 0; t < 4; ++t) {
    const int n  = nBase + t * 16 + l15;
    const float bv = bias[n];
    const int mrow = mBase + lh * 8;
#pragma unroll
    for (int j = 0; j < 8; ++j)
      gx[(size_t)(mrow + j) * FD + n] = __float2bfloat16(acc[t][j] + bv);
  }
}

// ---------------------------------------------------------------------------
// Per-layer scan init: zero double-buffered h-state (bf16) + grid barrier ctr.
// ---------------------------------------------------------------------------
__global__ void scan_init_kernel(__hip_bfloat16* __restrict__ hstate,
                                 unsigned* __restrict__ bar) {
  int i = blockIdx.x * 256 + threadIdx.x;
  if (i < 2 * BB * DD) hstate[i] = __float2bfloat16(0.f);
  if (i == 0) bar[0] = 0u;
}

// ---------------------------------------------------------------------------
// sLSTM sequential scan, one layer. 16 WGs x 128 threads (4 waves).
// WG w owns h-columns [w*16, w*16+16); gate rows g*256 + w*16 + j of Wh^T.
// Wh slice (64 rows x 256 K bf16 = 32KB) preloaded to LDS via TDM tensor
// loads (one 2D descriptor per gate). Each step:
//   wave g computes r[b, gate g cols] with 8x v_wmma_f32_16x16x32_bf16,
//   results staged in LDS, pointwise exp-gating in f32, new h written to the
//   double-buffered global h-state (bf16), residual accumulated into h-big,
//   then a monotonic atomic grid barrier across the 16 WGs.
// ---------------------------------------------------------------------------
__global__ void __launch_bounds__(128, 1)
slstm_scan_kernel(const __hip_bfloat16* __restrict__ WhT,  // [1024, 256] layer slice
                  const __hip_bfloat16* __restrict__ gx,   // [B*S, 1024]
                  __hip_bfloat16* __restrict__ hstate,     // [2][16][256]
                  float* __restrict__ hbig,                // [B,S,256] residual io
                  unsigned* __restrict__ bar) {
  __shared__ __hip_bfloat16 whs[64 * 256];   // 32KB: local row = g*16 + j
  __shared__ float rbuf[BB * 64];            // 4KB : r[b][g*16 + dd]

  const int w    = blockIdx.x;       // 0..15
  const int t    = threadIdx.x;      // 0..127
  const int lane = t & 31;
  const int wv   = t >> 5;           // wave == gate index 0..3
  const int l15  = lane & 15, lh = lane >> 4;

  // ---- TDM preload of this WG's Wh slice into LDS (wave 0 issues) ----
  if (wv == 0) {
    const uint32_t ldsBase = (uint32_t)(uintptr_t)(&whs[0]);
    for (int g = 0; g < 4; ++g) {
      uint64_t ga = (uint64_t)(uintptr_t)(WhT + ((size_t)g * DD + w * 16) * DD);
      tdm_v4u g0;
      g0[0] = 1u;                                      // count=1, user mode
      g0[1] = ldsBase + (uint32_t)(g * 16 * DD * 2);   // lds_addr (bytes)
      g0[2] = (uint32_t)ga;                            // global_addr[31:0]
      g0[3] = (uint32_t)((ga >> 32) & 0x1FFFFFFu) | (2u << 30);  // [56:32], type=2
      tdm_v8i g1;
      g1[0] = (1 << 16);        // data_size = 2 bytes (bf16)
      g1[1] = (256 << 16);      // tensor_dim0[15:0] = 256 (K)
      g1[2] = (16 << 16);       // tensor_dim1[15:0] = 16 rows
      g1[3] = (256 << 16);      // tile_dim0 = 256
      g1[4] = 16;               // tile_dim1 = 16, tile_dim2 = 0
      g1[5] = 256;              // tensor_dim0_stride = 256 elems
      g1[6] = 0;
      g1[7] = 0;
      tdm_v4i g2 = {0, 0, 0, 0};
      tdm_v4i g3 = {0, 0, 0, 0};
      tdm_v8i g4 = {0, 0, 0, 0, 0, 0, 0, 0};   // 6-arg form: extra group (unused, 2D tile)
      __builtin_amdgcn_tensor_load_to_lds(g0, g1, g2, g3, g4, 0);
    }
    __builtin_amdgcn_s_wait_tensorcnt(0);
  }
  __syncthreads();

  // per-thread state: 2 (b,dd) pairs
  const int p0 = t,        b0 = p0 >> 4, dd0 = p0 & 15;
  const int p1 = 128 + t,  b1 = p1 >> 4, dd1 = p1 & 15;
  float c0 = 0.f, n0 = 0.f, m0 = 0.f;
  float c1 = 0.f, n1 = 0.f, m1 = 0.f;

  unsigned target = 16u;
  for (int s = 0; s < SS; ++s) {
    const __hip_bfloat16* hin  = hstate + (size_t)(s & 1) * (BB * DD);
    __hip_bfloat16*       hout = hstate + (size_t)((s + 1) & 1) * (BB * DD);

    // ---- recurrent GEMM: wave wv -> gate wv, 16 cols, all 16 batches ----
    v8f acc = (v8f){0, 0, 0, 0, 0, 0, 0, 0};
    const __hip_bfloat16* aRow = hin + (size_t)l15 * DD + lh * 8;  // b = l15
#pragma unroll
    for (int kt = 0; kt < 8; ++kt) {
      const int k = kt * 32;
      BPack a, bm;
      a.u[0] = *(const uint4*)(aRow + k);
      a.u[1] = *(const uint4*)(aRow + k + 16);
      const __hip_bfloat16* bRow = &whs[(wv * 16 + l15) * DD + k + lh * 8];
      bm.u[0] = *(const uint4*)(bRow);
      bm.u[1] = *(const uint4*)(bRow + 16);
      acc = __builtin_amdgcn_wmma_f32_16x16x32_bf16(
          false, a.v, false, bm.v, (short)0, acc, false, false);
    }
#pragma unroll
    for (int j = 0; j < 8; ++j)
      rbuf[(j + lh * 8) * 64 + wv * 16 + l15] = acc[j];
    __syncthreads();

    // ---- pointwise exp-gating for this thread's 2 (b,dd) pairs ----
    {
      const int bs[2] = {b0, b1};
      const int ds[2] = {dd0, dd1};
      float* cs[2] = {&c0, &c1};
      float* ns[2] = {&n0, &n1};
      float* ms[2] = {&m0, &m1};
#pragma unroll
      for (int q = 0; q < 2; ++q) {
        const int b = bs[q], dd = ds[q];
        const int col = w * 16 + dd;
        const size_t gxoff = ((size_t)b * SS + s) * FD;
        float i_t = rbuf[b * 64 + dd]      + __bfloat162float(gx[gxoff + col]);
        float f_t = rbuf[b * 64 + 16 + dd] + __bfloat162float(gx[gxoff + 256 + col]);
        float z_t = rbuf[b * 64 + 32 + dd] + __bfloat162float(gx[gxoff + 512 + col]);
        float o_t = rbuf[b * 64 + 48 + dd] + __bfloat162float(gx[gxoff + 768 + col]);
        if (s + 1 < SS)
          __builtin_prefetch((const void*)(gx + gxoff + FD + col), 0, 1);
        float mm = *ms[q];
        float mn = fmaxf(f_t + mm, i_t);
        float ip = __expf(i_t - mn);
        float fp = __expf(f_t + mm - mn);
        float e2 = __expf(2.f * z_t);
        float tz = (e2 - 1.f) / (e2 + 1.f);          // tanh(z)
        float c  = fp * (*cs[q]) + ip * tz;
        float nn = fp * (*ns[q]) + ip;
        *cs[q] = c; *ns[q] = nn; *ms[q] = mn;
        float sig = 1.f / (1.f + __expf(-o_t));
        float hv  = sig * c / fmaxf(fabsf(nn), 1.f);
        hout[b * DD + col] = __float2bfloat16(hv);
        hbig[((size_t)b * SS + s) * DD + col] += hv;   // residual
      }
    }

    // ---- grid barrier across the 16 WGs (monotonic counter) ----
    __threadfence();
    __syncthreads();
    if (t == 0) {
      __hip_atomic_fetch_add(bar, 1u, __ATOMIC_RELEASE, __HIP_MEMORY_SCOPE_AGENT);
      while (__hip_atomic_load(bar, __ATOMIC_ACQUIRE, __HIP_MEMORY_SCOPE_AGENT) <
             target)
        __builtin_amdgcn_s_sleep(1);
    }
    __syncthreads();
    target += 16u;
  }
}

// ---------------------------------------------------------------------------
// Head: out[b] = h[b, S-1, :] . fc_W + fc_b
// ---------------------------------------------------------------------------
__global__ void head_kernel(const float* __restrict__ h,
                            const float* __restrict__ fcW,
                            const float* __restrict__ fcb,
                            float* __restrict__ out) {
  __shared__ float red[256];
  const int b = blockIdx.x;
  red[threadIdx.x] =
      h[((size_t)b * SS + (SS - 1)) * DD + threadIdx.x] * fcW[threadIdx.x];
  __syncthreads();
  for (int off = 128; off; off >>= 1) {
    if (threadIdx.x < off) red[threadIdx.x] += red[threadIdx.x + off];
    __syncthreads();
  }
  if (threadIdx.x == 0) out[b] = red[0] + fcb[0];
}

// ---------------------------------------------------------------------------
extern "C" void kernel_launch(void* const* d_in, const int* in_sizes, int n_in,
                              void* d_out, int out_size, void* d_ws,
                              size_t ws_size, hipStream_t stream) {
  const float* x    = (const float*)d_in[0];
  const float* tf   = (const float*)d_in[1];
  const float* in_W = (const float*)d_in[2];
  const float* in_b = (const float*)d_in[3];
  const float* ln_g = (const float*)d_in[4];
  const float* ln_b = (const float*)d_in[5];
  const float* Wx   = (const float*)d_in[6];
  const float* Wh   = (const float*)d_in[7];
  const float* bg   = (const float*)d_in[8];
  const float* fc_W = (const float*)d_in[9];
  const float* fc_b = (const float*)d_in[10];
  float* out = (float*)d_out;

  // workspace carve-up (~116 MB)
  char* ws = (char*)d_ws;
  size_t off = 0;
  float* h = (float*)(ws + off);                  off += (size_t)MROWS * DD * 4;
  __hip_bfloat16* xn  = (__hip_bfloat16*)(ws + off); off += (size_t)MROWS * DD * 2;
  __hip_bfloat16* gx  = (__hip_bfloat16*)(ws + off); off += (size_t)MROWS * FD * 2;
  __hip_bfloat16* WxT = (__hip_bfloat16*)(ws + off); off += (size_t)LL * FD * DD * 2;
  __hip_bfloat16* WhT = (__hip_bfloat16*)(ws + off); off += (size_t)LL * FD * DD * 2;
  __hip_bfloat16* hstate = (__hip_bfloat16*)(ws + off); off += (size_t)2 * BB * DD * 2;
  unsigned* bar = (unsigned*)(ws + off);          off += 256;

  prep_w_kernel<<<(LL * FD * DD) / 256, 256, 0, stream>>>(Wx, Wh, WxT, WhT);
  in_proj_kernel<<<(MROWS * DD) / 256, 256, 0, stream>>>(x, tf, in_W, in_b, h);

  for (int l = 0; l < LL; ++l) {
    ln_cast_kernel<<<MROWS / 8, 256, 0, stream>>>(h, ln_g + l * DD,
                                                  ln_b + l * DD, xn);
    dim3 gg(MROWS / 32, FD / 256);
    gemm_gx_kernel<<<gg, 256, 0, stream>>>(xn, WxT + (size_t)l * FD * DD,
                                           bg + l * FD, gx);
    scan_init_kernel<<<(2 * BB * DD) / 256, 256, 0, stream>>>(hstate, bar);
    slstm_scan_kernel<<<16, 128, 0, stream>>>(WhT + (size_t)l * FD * DD, gx,
                                              hstate, h, bar);
  }
  head_kernel<<<BB, 256, 0, stream>>>(h, fc_W, fc_b, out);
}
